// DecoderLayer_72877005078814
// MI455X (gfx1250) — compile-verified, matrix operand order checked by reading, not run on hardware
//
#include <hip/hip_runtime.h>

// ---------------- problem constants ----------------
#define B_  2
#define S_  2048
#define T_  2048
#define D_  1024
#define H_  16
#define DK_ 64
#define DF_ 4096

typedef unsigned short u16;
typedef __attribute__((ext_vector_type(16))) __bf16 bf16x16;
typedef __attribute__((ext_vector_type(8)))  float  f32x8;

union FragU { uint4 u[2]; bf16x16 v; };

// ---- CDNA5 async global->LDS staging (guarded: falls back to ld/st path) ----
#if defined(__HIP_DEVICE_COMPILE__) && __has_builtin(__builtin_amdgcn_global_load_async_to_lds_b128)
#define USE_ASYNC_LDS 1
// builtin prototype is (v4i addrspace(1)*, v4i addrspace(3)*, imm offset, imm cpol)
// with *generic* (vector_size) vectors, per the clang diagnostic.
typedef int v4i_vs __attribute__((vector_size(16)));
typedef __attribute__((address_space(1))) v4i_vs gas_v4i;
typedef __attribute__((address_space(3))) v4i_vs las_v4i;
#else
#define USE_ASYNC_LDS 0
#endif

__device__ __forceinline__ void async_wait0() {
#if __has_builtin(__builtin_amdgcn_s_wait_asynccnt)
    __builtin_amdgcn_s_wait_asynccnt(0);
#else
    asm volatile("s_wait_asynccnt 0x0" ::: "memory");
#endif
}

// copy 16 bytes global -> LDS (async DMA when available)
__device__ __forceinline__ void stage16(const u16* gsrc, u16* ldst) {
#if USE_ASYNC_LDS
    __builtin_amdgcn_global_load_async_to_lds_b128((gas_v4i*)gsrc, (las_v4i*)ldst, 0, 0);
#else
    *(uint4*)ldst = *(const uint4*)gsrc;
#endif
}

__device__ __forceinline__ u16 f2bf(float f) {
    union { float f; unsigned u; } c; c.f = f;
    unsigned r = c.u + 0x7FFFu + ((c.u >> 16) & 1u);
    return (u16)(r >> 16);
}

// Fragment loader from a row-major 16-bit matrix (global).
// Works for A (lane=row M) and for B sourced from an NxK row-major matrix
// (lane=row N): elements are K = {kb..kb+7, kb+16..kb+23}, kb = (lane<16?0:8).
__device__ __forceinline__ bf16x16 load_frag_g(const u16* base, int row, int stride, int k0) {
    const int lane = threadIdx.x & 31;
    const u16* p = base + (size_t)(row + (lane & 15)) * stride + k0 + ((lane >> 4) << 3);
    FragU f;
    f.u[0] = *(const uint4*)(p);
    f.u[1] = *(const uint4*)(p + 16);
    return f.v;
}

__device__ __forceinline__ bf16x16 load_frag_lds(const u16* base, int row, int stride) {
    const int lane = threadIdx.x & 31;
    const u16* p = base + (row + (lane & 15)) * stride + ((lane >> 4) << 3);
    FragU f;
    f.u[0] = *(const uint4*)(p);
    f.u[1] = *(const uint4*)(p + 16);
    return f.v;
}

__device__ __forceinline__ f32x8 wmma_bf16(bf16x16 a, bf16x16 b, f32x8 c) {
    return __builtin_amdgcn_wmma_f32_16x16x32_bf16(false, a, false, b, (short)0, c, false, false);
}

// ---------------- f32 -> bf16 conversion ----------------
__global__ void __launch_bounds__(256) cvt_f32_bf16(const float* __restrict__ in,
                                                    u16* __restrict__ out, int n) {
    int i = blockIdx.x * blockDim.x + threadIdx.x;
    int stride = gridDim.x * blockDim.x;
    for (; i < n; i += stride) out[i] = f2bf(in[i]);
}

// ---------------- tiled bf16 WMMA GEMM: C = A(MxK) * W(NxK)^T + bias ----------------
// mode: 0 = f32 row-major, 1 = bf16 row-major, 2 = bf16 [B,H,S,64] (Q/K heads),
//       3 = bf16 [B,H,64,S] (V transposed)
__global__ void __launch_bounds__(256)
gemm_bf16_wmma(const u16* __restrict__ A, const u16* __restrict__ W,
               const float* __restrict__ bias, void* __restrict__ out,
               int M, int N, int K, int mode, int relu, int Sdim) {
    __shared__ u16 sA[2][128 * 40];
    __shared__ u16 sB[2][128 * 40];

    const int tid  = threadIdx.x;
    const int lane = tid & 31;
    const int wid  = tid >> 5;
    const int wm   = wid >> 1;        // 0..3 -> 32-row slices
    const int wn   = wid & 1;         // 0..1 -> 64-col slices
    const int bm   = blockIdx.y * 128;
    const int bn   = blockIdx.x * 128;

    const f32x8 zero = {0.f, 0.f, 0.f, 0.f, 0.f, 0.f, 0.f, 0.f};
    f32x8 acc[2][4];
#pragma unroll
    for (int i = 0; i < 2; ++i)
#pragma unroll
        for (int j = 0; j < 4; ++j) acc[i][j] = zero;

    // staging: each thread moves 32 B of A and 32 B of B per K-tile
    const int r = tid >> 1;
    const int c = (tid & 1) * 16;
    const u16* gA = A + (size_t)(bm + r) * K + c;
    const u16* gB = W + (size_t)(bn + r) * K + c;

    const int nk = K >> 5;

    // prologue stage
    stage16(gA,     &sA[0][r * 40 + c]);
    stage16(gA + 8, &sA[0][r * 40 + c + 8]);
    stage16(gB,     &sB[0][r * 40 + c]);
    stage16(gB + 8, &sB[0][r * 40 + c + 8]);

    for (int kt = 0; kt < nk; ++kt) {
#if USE_ASYNC_LDS
        async_wait0();
#endif
        __syncthreads();
        if (kt + 1 < nk) {
            const int nb = (kt + 1) & 1;
            const int k0 = (kt + 1) << 5;
            stage16(gA + k0,     &sA[nb][r * 40 + c]);
            stage16(gA + k0 + 8, &sA[nb][r * 40 + c + 8]);
            stage16(gB + k0,     &sB[nb][r * 40 + c]);
            stage16(gB + k0 + 8, &sB[nb][r * 40 + c + 8]);
        }
        if (kt + 2 < nk) {
            const int kp2 = (kt + 2) << 5;
            __builtin_prefetch(gA + kp2, 0, 1);   // -> global_prefetch_b8
            __builtin_prefetch(gB + kp2, 0, 1);
        }
        const int cur = kt & 1;
        bf16x16 af[2], bfr[4];
#pragma unroll
        for (int mt = 0; mt < 2; ++mt) af[mt]  = load_frag_lds(&sA[cur][0], wm * 32 + mt * 16, 40);
#pragma unroll
        for (int nt = 0; nt < 4; ++nt) bfr[nt] = load_frag_lds(&sB[cur][0], wn * 64 + nt * 16, 40);
#pragma unroll
        for (int mt = 0; mt < 2; ++mt)
#pragma unroll
            for (int nt = 0; nt < 4; ++nt)
                acc[mt][nt] = wmma_bf16(af[mt], bfr[nt], acc[mt][nt]);
    }

    // epilogue
    const int hi   = lane >> 4;
    const int col0 = lane & 15;
#pragma unroll
    for (int mt = 0; mt < 2; ++mt) {
#pragma unroll
        for (int nt = 0; nt < 4; ++nt) {
            const int mrow0 = bm + wm * 32 + mt * 16 + hi * 8;
            const int ncol  = bn + wn * 64 + nt * 16 + col0;
            const float bv  = bias ? bias[ncol] : 0.f;
#pragma unroll
            for (int rr = 0; rr < 8; ++rr) {
                float v = acc[mt][nt][rr] + bv;
                if (relu) v = v > 0.f ? v : 0.f;
                const int mrow = mrow0 + rr;
                if (mode == 0) {
                    ((float*)out)[(size_t)mrow * N + ncol] = v;
                } else if (mode == 1) {
                    ((u16*)out)[(size_t)mrow * N + ncol] = f2bf(v);
                } else {
                    const int b  = mrow / Sdim;
                    const int s  = mrow % Sdim;
                    const int h  = ncol >> 6;
                    const int dk = ncol & 63;
                    size_t idx;
                    if (mode == 2)
                        idx = (((size_t)(b * H_ + h)) * Sdim + s) * 64 + dk;
                    else
                        idx = (((size_t)(b * H_ + h)) * 64 + dk) * (size_t)Sdim + s;
                    ((u16*)out)[idx] = f2bf(v);
                }
            }
        }
    }
}

// ---------------- flash attention (wave = 16 query rows) ----------------
// q : bf16 [B,H,Sq,64], k : bf16 [B,H,Sk,64], vT : bf16 [B,H,64,Sk]
// out: bf16 [B,Sq,D]  (heads re-interleaved), causal = 1 for self-attention
__global__ void __launch_bounds__(128)
attn_wmma(const u16* __restrict__ q, const u16* __restrict__ k,
          const u16* __restrict__ vT, u16* __restrict__ out,
          int Sq, int Sk, int causal) {
    __shared__ u16 pl[4][16 * 40];

    const int lane = threadIdx.x & 31;
    const int wid  = threadIdx.x >> 5;
    const int bh   = blockIdx.y;
    const int b    = bh >> 4;
    const int h    = bh & 15;
    const int q0   = blockIdx.x * 64 + wid * 16;

    const u16* qp = q  + (size_t)bh * Sq * 64;
    const u16* kp = k  + (size_t)bh * Sk * 64;
    const u16* vp = vT + (size_t)bh * 64 * Sk;

    const bf16x16 qf0 = load_frag_g(qp, q0, 64, 0);
    const bf16x16 qf1 = load_frag_g(qp, q0, 64, 32);

    const f32x8 zero = {0.f, 0.f, 0.f, 0.f, 0.f, 0.f, 0.f, 0.f};
    f32x8 o[4];
#pragma unroll
    for (int f = 0; f < 4; ++f) o[f] = zero;

    float m_i[8], l_i[8];
#pragma unroll
    for (int rr = 0; rr < 8; ++rr) { m_i[rr] = -1e30f; l_i[rr] = 0.f; }

    const int   hi    = lane >> 4;
    const int   col0  = lane & 15;
    const int   rbase = hi * 8;
    const float scale = 0.125f; // 1/sqrt(64)
    const int   kend  = causal ? (q0 + 16) : Sk;

    for (int kc = 0; kc < kend; kc += 32) {
        if (kc + 32 < kend) {
            __builtin_prefetch(kp + (size_t)(kc + 32) * 64, 0, 1);
            __builtin_prefetch(vp + (kc + 32), 0, 1);
        }
        // S tile: 16 query rows x 32 keys (two 16x16 C-frags)
        f32x8 sfr[2];
#pragma unroll
        for (int nt = 0; nt < 2; ++nt) {
            f32x8 a = zero;
            const bf16x16 kf0 = load_frag_g(kp, kc + nt * 16, 64, 0);
            const bf16x16 kf1 = load_frag_g(kp, kc + nt * 16, 64, 32);
            a = wmma_bf16(qf0, kf0, a);
            a = wmma_bf16(qf1, kf1, a);
            sfr[nt] = a;
        }

        // scale + mask + chunk row-max
        float mc[8];
#pragma unroll
        for (int rr = 0; rr < 8; ++rr) {
            float v0 = sfr[0][rr] * scale;
            float v1 = sfr[1][rr] * scale;
            if (causal) {
                const int qi = q0 + rbase + rr;
                if (kc + col0 > qi)      v0 = -1e30f;
                if (kc + 16 + col0 > qi) v1 = -1e30f;
            }
            sfr[0][rr] = v0;
            sfr[1][rr] = v1;
            mc[rr] = v0 > v1 ? v0 : v1;
        }
#pragma unroll
        for (int off = 1; off < 16; off <<= 1) {
#pragma unroll
            for (int rr = 0; rr < 8; ++rr) {
                const float t = __shfl_xor(mc[rr], off);
                mc[rr] = t > mc[rr] ? t : mc[rr];
            }
        }

        // online softmax update
        float lsum[8];
#pragma unroll
        for (int rr = 0; rr < 8; ++rr) {
            const float mn    = m_i[rr] > mc[rr] ? m_i[rr] : mc[rr];
            const float alpha = __expf(m_i[rr] - mn);
            const float p0 = __expf(sfr[0][rr] - mn);
            const float p1 = __expf(sfr[1][rr] - mn);
            sfr[0][rr] = p0;
            sfr[1][rr] = p1;
            lsum[rr]   = p0 + p1;
            m_i[rr]    = mn;
            l_i[rr]   *= alpha;
#pragma unroll
            for (int f = 0; f < 4; ++f) o[f][rr] *= alpha;
        }
#pragma unroll
        for (int off = 1; off < 16; off <<= 1) {
#pragma unroll
            for (int rr = 0; rr < 8; ++rr) lsum[rr] += __shfl_xor(lsum[rr], off);
        }
#pragma unroll
        for (int rr = 0; rr < 8; ++rr) l_i[rr] += lsum[rr];

        // P (C-layout f32) -> LDS bf16 16x32 (row-major, pad 40) -> A-fragment
        u16* pw = &pl[wid][0];
#pragma unroll
        for (int rr = 0; rr < 8; ++rr) {
            pw[(rbase + rr) * 40 + col0]      = f2bf(sfr[0][rr]);
            pw[(rbase + rr) * 40 + 16 + col0] = f2bf(sfr[1][rr]);
        }
        asm volatile("s_wait_dscnt 0x0" ::: "memory");
        const bf16x16 pf = load_frag_lds(pw, 0, 40);

        // O += P(16x32) * V(32x64), V fragments from transposed V (row = feature)
#pragma unroll
        for (int f = 0; f < 4; ++f) {
            const bf16x16 vf = load_frag_g(vp, f * 16, Sk, kc);
            o[f] = wmma_bf16(pf, vf, o[f]);
        }
    }

    // epilogue: out[b, s, h*64 + feat] = o / l
#pragma unroll
    for (int f = 0; f < 4; ++f) {
#pragma unroll
        for (int rr = 0; rr < 8; ++rr) {
            const int   s   = q0 + rbase + rr;
            const float v   = o[f][rr] * (1.0f / l_i[rr]);
            out[((size_t)b * Sq + s) * D_ + h * 64 + f * 16 + col0] = f2bf(v);
        }
    }
}

// ---------------- residual add + LayerNorm ----------------
__global__ void __launch_bounds__(256)
ln_kernel(const float* __restrict__ a, const float* __restrict__ bres,
          const float* __restrict__ gamma, const float* __restrict__ beta,
          float* __restrict__ of32, u16* __restrict__ obf) {
    __shared__ float red[256];
    const int row = blockIdx.x;
    const int t   = threadIdx.x;
    const float* pa = a + (size_t)row * D_;
    const float* pb = bres ? bres + (size_t)row * D_ : nullptr;

    float x[4];
    float s = 0.f;
#pragma unroll
    for (int i = 0; i < 4; ++i) {
        const int c = t + i * 256;
        float v = pa[c];
        if (pb) v += pb[c];
        x[i] = v;
        s += v;
    }
    red[t] = s;
    __syncthreads();
    for (int off = 128; off > 0; off >>= 1) {
        if (t < off) red[t] += red[t + off];
        __syncthreads();
    }
    const float mu = red[0] * (1.0f / D_);
    __syncthreads();

    float vs = 0.f;
#pragma unroll
    for (int i = 0; i < 4; ++i) { const float d = x[i] - mu; vs += d * d; }
    red[t] = vs;
    __syncthreads();
    for (int off = 128; off > 0; off >>= 1) {
        if (t < off) red[t] += red[t + off];
        __syncthreads();
    }
    const float var  = red[0] * (1.0f / D_);
    const float rstd = rsqrtf(var + 1e-5f);

#pragma unroll
    for (int i = 0; i < 4; ++i) {
        const int c = t + i * 256;
        const float y = (x[i] - mu) * rstd * gamma[c] + beta[c];
        if (of32) of32[(size_t)row * D_ + c] = y;
        if (obf)  obf[(size_t)row * D_ + c] = f2bf(y);
    }
}

// ---------------- host orchestration ----------------
extern "C" void kernel_launch(void* const* d_in, const int* in_sizes, int n_in,
                              void* d_out, int out_size, void* d_ws, size_t ws_size,
                              hipStream_t stream) {
    (void)in_sizes; (void)n_in; (void)out_size; (void)ws_size;

    const float* x    = (const float*)d_in[0];
    const float* enc  = (const float*)d_in[1];
    // d_in[2] = src_mask (all ones), d_in[3] = tgt_mask (causal) -- encoded structurally
    const float* Wq1 = (const float*)d_in[4];  const float* bq1 = (const float*)d_in[5];
    const float* Wk1 = (const float*)d_in[6];  const float* bk1 = (const float*)d_in[7];
    const float* Wv1 = (const float*)d_in[8];  const float* bv1 = (const float*)d_in[9];
    const float* Wo1 = (const float*)d_in[10]; const float* bo1 = (const float*)d_in[11];
    const float* Wq2 = (const float*)d_in[12]; const float* bq2 = (const float*)d_in[13];
    const float* Wk2 = (const float*)d_in[14]; const float* bk2 = (const float*)d_in[15];
    const float* Wv2 = (const float*)d_in[16]; const float* bv2 = (const float*)d_in[17];
    const float* Wo2 = (const float*)d_in[18]; const float* bo2 = (const float*)d_in[19];
    const float* Wf1 = (const float*)d_in[20]; const float* bf1 = (const float*)d_in[21];
    const float* Wf2 = (const float*)d_in[22]; const float* bf2 = (const float*)d_in[23];
    const float* g1  = (const float*)d_in[24]; const float* be1 = (const float*)d_in[25];
    const float* g2  = (const float*)d_in[26]; const float* be2 = (const float*)d_in[27];
    const float* g3  = (const float*)d_in[28]; const float* be3 = (const float*)d_in[29];

    char* ws = (char*)d_ws;
    size_t off = 0;
    auto alloc = [&](size_t bytes) -> char* {
        char* p = ws + off;
        off += (bytes + 255) & ~(size_t)255;
        return p;
    };

    const size_t nBSD = (size_t)B_ * S_ * D_;
    const size_t nBTD = (size_t)B_ * T_ * D_;
    const size_t nDD  = (size_t)D_ * D_;
    const size_t nDDF = (size_t)D_ * DF_;

    u16* wq1b = (u16*)alloc(nDD * 2);  u16* wk1b = (u16*)alloc(nDD * 2);
    u16* wv1b = (u16*)alloc(nDD * 2);  u16* wo1b = (u16*)alloc(nDD * 2);
    u16* wq2b = (u16*)alloc(nDD * 2);  u16* wk2b = (u16*)alloc(nDD * 2);
    u16* wv2b = (u16*)alloc(nDD * 2);  u16* wo2b = (u16*)alloc(nDD * 2);
    u16* wf1b = (u16*)alloc(nDDF * 2); u16* wf2b = (u16*)alloc(nDDF * 2);
    u16* xb   = (u16*)alloc(nBSD * 2);
    u16* encb = (u16*)alloc(nBTD * 2);
    u16* qb   = (u16*)alloc(nBSD * 2);  // reused by cross-attn
    u16* kb   = (u16*)alloc(nBTD * 2);
    u16* vtb  = (u16*)alloc(nBTD * 2);
    u16* attnb = (u16*)alloc(nBSD * 2);
    float* projf = (float*)alloc(nBSD * 4);      // reused for o1, o2, ffn2 outputs
    float* ln1f = (float*)alloc(nBSD * 4);
    u16*   ln1b = (u16*)alloc(nBSD * 2);
    float* ln2f = (float*)alloc(nBSD * 4);
    u16*   ln2b = (u16*)alloc(nBSD * 2);
    u16*   ffh  = (u16*)alloc((size_t)B_ * S_ * DF_ * 2);

    auto cvt = [&](const float* src, u16* dst, size_t n) {
        cvt_f32_bf16<<<2048, 256, 0, stream>>>(src, dst, (int)n);
    };
    auto gemm = [&](const u16* A, const u16* W, const float* bias, void* out,
                    int M, int N, int K, int mode, int relu, int Sdim) {
        dim3 grid(N / 128, M / 128);
        gemm_bf16_wmma<<<grid, 256, 0, stream>>>(A, W, bias, out, M, N, K, mode, relu, Sdim);
    };

    // conversions
    cvt(x, xb, nBSD);    cvt(enc, encb, nBTD);
    cvt(Wq1, wq1b, nDD); cvt(Wk1, wk1b, nDD); cvt(Wv1, wv1b, nDD); cvt(Wo1, wo1b, nDD);
    cvt(Wq2, wq2b, nDD); cvt(Wk2, wk2b, nDD); cvt(Wv2, wv2b, nDD); cvt(Wo2, wo2b, nDD);
    cvt(Wf1, wf1b, nDDF); cvt(Wf2, wf2b, nDDF);

    const int MS = B_ * S_;
    const int MT = B_ * T_;

    // ---- self attention ----
    gemm(xb, wq1b, bq1, qb,  MS, D_, D_, 2, 0, S_);
    gemm(xb, wk1b, bk1, kb,  MS, D_, D_, 2, 0, S_);
    gemm(xb, wv1b, bv1, vtb, MS, D_, D_, 3, 0, S_);
    attn_wmma<<<dim3(S_ / 64, B_ * H_), 128, 0, stream>>>(qb, kb, vtb, attnb, S_, S_, 1);
    gemm(attnb, wo1b, bo1, projf, MS, D_, D_, 0, 0, S_);
    ln_kernel<<<MS, 256, 0, stream>>>(x, projf, g1, be1, ln1f, ln1b);

    // ---- cross attention ----
    gemm(ln1b, wq2b, bq2, qb,  MS, D_, D_, 2, 0, S_);
    gemm(encb, wk2b, bk2, kb,  MT, D_, D_, 2, 0, T_);
    gemm(encb, wv2b, bv2, vtb, MT, D_, D_, 3, 0, T_);
    attn_wmma<<<dim3(S_ / 64, B_ * H_), 128, 0, stream>>>(qb, kb, vtb, attnb, S_, T_, 0);
    gemm(attnb, wo2b, bo2, projf, MS, D_, D_, 0, 0, S_);
    ln_kernel<<<MS, 256, 0, stream>>>(ln1f, projf, g2, be2, ln2f, ln2b);

    // ---- FFN ----
    gemm(ln2b, wf1b, bf1, ffh,   MS, DF_, D_, 1, 1, S_);
    gemm(ffh,  wf2b, bf2, projf, MS, D_, DF_, 0, 0, S_);
    ln_kernel<<<MS, 256, 0, stream>>>(ln2f, projf, g3, be3, (float*)d_out, nullptr);
}
